// TransformerBlock_20933670601475
// MI455X (gfx1250) — matrix operand order repaired and checked by reading, not compiled, not run on hardware
//
#include <hip/hip_runtime.h>
#include <hip/hip_bf16.h>
#include <math.h>

// ---------------- problem dims (compile-time) ----------------
#define EMBED  1024
#define NHEADS 16
#define HDIM   64
#define FFDIM  4096
#define TSEQ   2048
#define NBATCH 2
#define BTOK   (NBATCH * TSEQ)   // 4096 rows
#define LN_EPS 1e-5f

typedef __bf16 bf16_t;
typedef __attribute__((ext_vector_type(16))) __bf16 v16bf;
typedef __attribute__((ext_vector_type(8)))  __bf16 v8bf;
typedef __attribute__((ext_vector_type(8)))  float  v8f;
typedef __attribute__((ext_vector_type(4)))  int    v4i;

#define GLOBAL_AS __attribute__((address_space(1)))
#define LDS_AS    __attribute__((address_space(3)))

#if __has_builtin(__builtin_amdgcn_global_load_async_to_lds_b128) && \
    __has_builtin(__builtin_amdgcn_s_wait_asynccnt)
#define USE_ASYNC_LDS 1
#else
#define USE_ASYNC_LDS 0
#endif

// ---------------- WMMA helpers ----------------
__device__ __forceinline__ v8f zero_v8f() {
    v8f z;
#pragma unroll
    for (int i = 0; i < 8; ++i) z[i] = 0.f;
    return z;
}

__device__ __forceinline__ v8f wmma_bf16(v16bf a, v16bf b, v8f c) {
    // D = A(16x32 bf16) * B(32x16 bf16) + C(16x16 f32)
    return __builtin_amdgcn_wmma_f32_16x16x32_bf16(
        /*neg_a=*/false, a, /*neg_b=*/false, b,
        /*c_mod=*/(short)0, c, /*reuse_a=*/false, /*reuse_b=*/false);
}

__device__ __forceinline__ v16bf cat8(v8bf lo, v8bf hi) {
    v16bf r;
#pragma unroll
    for (int j = 0; j < 8; ++j) { r[j] = lo[j]; r[8 + j] = hi[j]; }
    return r;
}

// A-fragment (16x32 bf16, row-major source, leading dim `ld` elements).
// ISA layout: lane l holds row (l&15); VGPR0-3 <- K = (l>>4)*8 .. +7,
// VGPR4-7 <- K = (l>>4)*8 + 16 .. +23.  base must be 16B aligned, ld%8==0.
__device__ __forceinline__ v16bf load_a_frag(const bf16_t* base, int ld) {
    const int lane = threadIdx.x & 31;
    const bf16_t* p = base + (size_t)(lane & 15) * ld + ((lane >> 4) * 8);
    v8bf lo = *(const v8bf*)(p);
    v8bf hi = *(const v8bf*)(p + 16);
    return cat8(lo, hi);
}

// B-fragment (32x16 bf16) from a TRANSPOSED source Bt[n][k] (ld elements).
// ISA layout: lane l holds column (l&15); VGPR0-3 <- K = (l>>4)*8 .. +7,
// VGPR4-7 <- K = (l>>4)*8 + 16 .. +23.
__device__ __forceinline__ v16bf load_b_frag_t(const bf16_t* baseT, int ld) {
    const int lane = threadIdx.x & 31;
    const bf16_t* p = baseT + (size_t)(lane & 15) * ld + ((lane >> 4) * 8);
    v8bf lo = *(const v8bf*)(p);
    v8bf hi = *(const v8bf*)(p + 16);
    return cat8(lo, hi);
}

#if USE_ASYNC_LDS
// CDNA5 async copy: 16B per lane, global -> LDS, tracked by ASYNCcnt.
// Param 0 is int4* in addrspace(1) (prints as "__device__"); param 1 is the
// LDS destination in addrspace(3).
__device__ __forceinline__ void async_copy16(const bf16_t* g, bf16_t* l) {
    __builtin_amdgcn_global_load_async_to_lds_b128(
        (GLOBAL_AS v4i*)g, (LDS_AS v4i*)l, 0, 0);
}
#endif

__device__ __forceinline__ float gelu_exact(float v) {
    return 0.5f * v * (1.0f + erff(v * 0.70710678118654752f));
}

// ---------------- weight convert + transpose (fp32 W[K][N] -> bf16 Wt[N][K]) ----
__global__ void cvt_transpose_kernel(const float* __restrict__ W,
                                     bf16_t* __restrict__ Wt,
                                     int K, int N) {
    size_t total = (size_t)K * N;
    for (size_t idx = (size_t)blockIdx.x * blockDim.x + threadIdx.x;
         idx < total; idx += (size_t)gridDim.x * blockDim.x) {
        size_t n = idx / K, k = idx % K;
        Wt[idx] = (bf16_t)W[k * (size_t)N + n];
    }
}

// ---------------- LayerNorm (fp32 in -> bf16 out), one block per row ----------
__global__ __launch_bounds__(256) void ln_bf16_kernel(const float* __restrict__ x,
                                                      const float* __restrict__ g,
                                                      const float* __restrict__ b,
                                                      bf16_t* __restrict__ out) {
    const int row = blockIdx.x;
    const float* xr = x + (size_t)row * EMBED;
    float s = 0.f, ss = 0.f;
    for (int i = threadIdx.x; i < EMBED; i += 256) {
        float v = xr[i]; s += v; ss += v * v;
    }
    __shared__ float rs[256], rq[256];
    rs[threadIdx.x] = s; rq[threadIdx.x] = ss;
    __syncthreads();
    for (int st = 128; st > 0; st >>= 1) {
        if (threadIdx.x < st) {
            rs[threadIdx.x] += rs[threadIdx.x + st];
            rq[threadIdx.x] += rq[threadIdx.x + st];
        }
        __syncthreads();
    }
    const float mu  = rs[0] * (1.0f / EMBED);
    const float var = rq[0] * (1.0f / EMBED) - mu * mu;
    const float inv = rsqrtf(var + LN_EPS);
    bf16_t* orow = out + (size_t)row * EMBED;
    for (int i = threadIdx.x; i < EMBED; i += 256)
        orow[i] = (bf16_t)((xr[i] - mu) * inv * g[i] + b[i]);
}

// ---------------- tiled WMMA GEMM: C = A(MxK bf16) @ B(KxN, given as Bt[N][K]) --
// block: 256 threads (8 waves); macro tile 64(M) x 128(N); waves in 2x4 grid,
// each wave computes a 32x32 patch = 2x2 WMMA tiles (4 accumulators).
// K staged 32-wide, double-buffered LDS (async-to-LDS when available).
// mode 0: out_bf16 = acc + bias
// mode 1: out_bf16 = gelu(acc + bias)
// mode 2: out_f32  = acc + bias + resid[row][col]
__global__ __launch_bounds__(256) void gemm_bf16_wmma_kernel(
    const bf16_t* __restrict__ A, const bf16_t* __restrict__ Bt,
    const float* __restrict__ bias, const float* __restrict__ resid,
    void* __restrict__ outPtr, int M, int N, int K, int mode) {
    (void)M;
    __shared__ bf16_t As[2][64][40];    // 64 rows x 32 K (+8 pad; 80B pitch)
    __shared__ bf16_t BsT[2][128][40];  // 128 cols x 32 K (+8 pad)

    const int w    = threadIdx.x >> 5;
    const int lane = threadIdx.x & 31;
    const int rowBase = blockIdx.y * 64;
    const int colBase = blockIdx.x * 128;
    const int mw = (w >> 2) * 32;       // wave's M patch (0 or 32)
    const int nw = (w & 3) * 32;        // wave's N patch (0..96)

    // staging assignment: A tile 64x32 -> one b128/thread; B tile 128x32 -> two
    const int ar = threadIdx.x >> 2, ac = (threadIdx.x & 3) * 8;
    const int bn = threadIdx.x >> 1, bc = (threadIdx.x & 1) * 16;
    const bf16_t* agp = A  + (size_t)(rowBase + ar) * K + ac;
    const bf16_t* bgp = Bt + (size_t)(colBase + bn) * K + bc;

    v8f acc00 = zero_v8f(), acc01 = zero_v8f();
    v8f acc10 = zero_v8f(), acc11 = zero_v8f();

    const int nsteps = K / 32;
    int cur = 0;

#if USE_ASYNC_LDS
    async_copy16(agp,     &As[0][ar][ac]);
    async_copy16(bgp,     &BsT[0][bn][bc]);
    async_copy16(bgp + 8, &BsT[0][bn][bc + 8]);
    __builtin_amdgcn_s_wait_asynccnt(0);
    __syncthreads();
    for (int s = 0; s < nsteps; ++s) {
        if (s + 1 < nsteps) {           // prefetch next tile into other buffer
            const int k = (s + 1) * 32;
            async_copy16(agp + k,     &As[cur ^ 1][ar][ac]);
            async_copy16(bgp + k,     &BsT[cur ^ 1][bn][bc]);
            async_copy16(bgp + k + 8, &BsT[cur ^ 1][bn][bc + 8]);
        }
        v16bf a0 = load_a_frag(&As[cur][mw][0], 40);
        v16bf a1 = load_a_frag(&As[cur][mw + 16][0], 40);
        v16bf b0 = load_b_frag_t(&BsT[cur][nw][0], 40);
        v16bf b1 = load_b_frag_t(&BsT[cur][nw + 16][0], 40);
        acc00 = wmma_bf16(a0, b0, acc00);
        acc01 = wmma_bf16(a0, b1, acc01);
        acc10 = wmma_bf16(a1, b0, acc10);
        acc11 = wmma_bf16(a1, b1, acc11);
        __builtin_amdgcn_s_wait_asynccnt(0);
        __syncthreads();
        cur ^= 1;
    }
#else
    uint4 pa, pb0, pb1;
    pa  = *(const uint4*)(agp);
    pb0 = *(const uint4*)(bgp);
    pb1 = *(const uint4*)(bgp + 8);
    *(uint4*)&As[0][ar][ac]       = pa;
    *(uint4*)&BsT[0][bn][bc]      = pb0;
    *(uint4*)&BsT[0][bn][bc + 8]  = pb1;
    __syncthreads();
    for (int s = 0; s < nsteps; ++s) {
        const bool nxt = (s + 1 < nsteps);
        if (nxt) {                      // global loads overlap compute below
            const int k = (s + 1) * 32;
            pa  = *(const uint4*)(agp + k);
            pb0 = *(const uint4*)(bgp + k);
            pb1 = *(const uint4*)(bgp + k + 8);
        }
        v16bf a0 = load_a_frag(&As[cur][mw][0], 40);
        v16bf a1 = load_a_frag(&As[cur][mw + 16][0], 40);
        v16bf b0 = load_b_frag_t(&BsT[cur][nw][0], 40);
        v16bf b1 = load_b_frag_t(&BsT[cur][nw + 16][0], 40);
        acc00 = wmma_bf16(a0, b0, acc00);
        acc01 = wmma_bf16(a0, b1, acc01);
        acc10 = wmma_bf16(a1, b0, acc10);
        acc11 = wmma_bf16(a1, b1, acc11);
        if (nxt) {                      // write NEXT buffer (disjoint from reads)
            *(uint4*)&As[cur ^ 1][ar][ac]      = pa;
            *(uint4*)&BsT[cur ^ 1][bn][bc]     = pb0;
            *(uint4*)&BsT[cur ^ 1][bn][bc + 8] = pb1;
        }
        __syncthreads();
        cur ^= 1;
    }
#endif

    // epilogue: 4 tiles per wave
    const int n  = lane & 15;
    const int hi = lane >> 4;
    auto store_tile = [&](v8f acc, int rBase, int cBase) {
        const int col = cBase + n;
        const float bcol = bias ? bias[col] : 0.f;
#pragma unroll
        for (int v = 0; v < 8; ++v) {
            const int row = rBase + v + 8 * hi;
            float val = acc[v] + bcol;
            if (mode == 1) val = gelu_exact(val);
            if (mode == 2) {
                ((float*)outPtr)[(size_t)row * N + col] =
                    val + resid[(size_t)row * N + col];
            } else {
                ((bf16_t*)outPtr)[(size_t)row * N + col] = (bf16_t)val;
            }
        }
    };
    store_tile(acc00, rowBase + mw,      colBase + nw);
    store_tile(acc01, rowBase + mw,      colBase + nw + 16);
    store_tile(acc10, rowBase + mw + 16, colBase + nw);
    store_tile(acc11, rowBase + mw + 16, colBase + nw + 16);
}

// ---------------- flash attention (causal), bf16 WMMA --------------------------
// grid: (T/64, NHEADS, NBATCH); block: 128 threads = 4 waves; wave w -> 16 q-rows.
__global__ __launch_bounds__(128) void attention_wmma_kernel(
    const bf16_t* __restrict__ qkv,   // [B*T][3*EMBED], q|k|v per head inside
    bf16_t* __restrict__ attn_out) {  // [B*T][EMBED]
    const int h    = blockIdx.y;
    const int bb   = blockIdx.z;
    const int w    = threadIdx.x >> 5;
    const int lane = threadIdx.x & 31;
    const int n    = lane & 15;
    const int hi   = lane >> 4;
    const int qBase = blockIdx.x * 64;
    const int q0    = qBase + w * 16;
    const int ldq   = 3 * EMBED;
    const float scale = 0.125f;       // 1/sqrt(64)

    __shared__ bf16_t Ks[32][72];     // 32 keys x 64 dims (+8 pad) (row-major = Bt form)
    __shared__ bf16_t Vt[64][40];     // 64 dims x 32 keys (+8 pad) (transposed V)
    __shared__ bf16_t Ps[4][16][40];  // per-wave P tile: 16 q x 32 keys (+8 pad)

    // load + pre-scale Q fragments (D=64 -> two K=32 chunks), kept in registers
    const bf16_t* qbase = qkv + ((size_t)bb * TSEQ + q0) * ldq + h * HDIM;
    v16bf aQ0 = load_a_frag(qbase, ldq);
    v16bf aQ1 = load_a_frag(qbase + 32, ldq);
#pragma unroll
    for (int j = 0; j < 16; ++j) {
        aQ0[j] = (bf16_t)((float)aQ0[j] * scale);
        aQ1[j] = (bf16_t)((float)aQ1[j] * scale);
    }

    v8f Oacc[4] = {zero_v8f(), zero_v8f(), zero_v8f(), zero_v8f()};
    float mrow[8], lrow[8];
#pragma unroll
    for (int v = 0; v < 8; ++v) { mrow[v] = -1e30f; lrow[v] = 0.f; }

    const int nBlocks = (qBase + 64) / 32;
    for (int blk = 0; blk < nBlocks; ++blk) {
        const int j0 = blk * 32;
        __syncthreads();              // previous iteration done with Ks/Vt
        {                             // stage K row-major and V transposed
            const int t   = threadIdx.x;
            const int key = t >> 2;
            const int c   = (t & 3) * 16;
            const bf16_t* krow = qkv + ((size_t)bb * TSEQ + j0 + key) * ldq
                                 + EMBED + h * HDIM + c;
            *(uint4*)&Ks[key][c]     = *(const uint4*)(krow);
            *(uint4*)&Ks[key][c + 8] = *(const uint4*)(krow + 8);
            const bf16_t* vrow = qkv + ((size_t)bb * TSEQ + j0 + key) * ldq
                                 + 2 * EMBED + h * HDIM + c;
            bf16_t tmp[16];
            *(uint4*)&tmp[0] = *(const uint4*)(vrow);
            *(uint4*)&tmp[8] = *(const uint4*)(vrow + 8);
#pragma unroll
            for (int i2 = 0; i2 < 16; ++i2) Vt[c + i2][key] = tmp[i2];
        }
        __syncthreads();

        const bool active = (j0 <= q0 + 15);  // wave-uniform
        if (active) {
            // S = Q @ K^T for two 16-key sub-tiles
            v8f s0 = zero_v8f(), s1 = zero_v8f();
            s0 = wmma_bf16(aQ0, load_b_frag_t(&Ks[0][0], 72), s0);
            s0 = wmma_bf16(aQ1, load_b_frag_t(&Ks[0][32], 72), s0);
            s1 = wmma_bf16(aQ0, load_b_frag_t(&Ks[16][0], 72), s1);
            s1 = wmma_bf16(aQ1, load_b_frag_t(&Ks[16][32], 72), s1);

            float p0[8], p1[8];
#pragma unroll
            for (int v = 0; v < 8; ++v) {        // causal mask
                const int qrow = q0 + v + 8 * hi;
                p0[v] = (j0 + n > qrow)      ? -1e30f : s0[v];
                p1[v] = (j0 + 16 + n > qrow) ? -1e30f : s1[v];
            }
#pragma unroll
            for (int v = 0; v < 8; ++v) {        // online softmax update
                float rm = fmaxf(p0[v], p1[v]);
#pragma unroll
                for (int msk = 1; msk < 16; msk <<= 1)
                    rm = fmaxf(rm, __shfl_xor(rm, msk, 32));
                const float mn  = fmaxf(mrow[v], rm);
                const float fac = __expf(mrow[v] - mn);
                mrow[v] = mn;
                p0[v] = __expf(p0[v] - mn);
                p1[v] = __expf(p1[v] - mn);
                float rsum = p0[v] + p1[v];
#pragma unroll
                for (int msk = 1; msk < 16; msk <<= 1)
                    rsum += __shfl_xor(rsum, msk, 32);
                lrow[v] = lrow[v] * fac + rsum;
#pragma unroll
                for (int t2 = 0; t2 < 4; ++t2) Oacc[t2][v] *= fac;
            }
            // write P tile (C-layout -> LDS, re-read in A-layout)
#pragma unroll
            for (int v = 0; v < 8; ++v) {
                Ps[w][v + 8 * hi][n]      = (bf16_t)p0[v];
                Ps[w][v + 8 * hi][16 + n] = (bf16_t)p1[v];
            }
        }
        __syncthreads();              // also orders same-wave P RAW safely
        if (active) {
            v16bf aP = load_a_frag(&Ps[w][0][0], 40);
#pragma unroll
            for (int t2 = 0; t2 < 4; ++t2)
                Oacc[t2] = wmma_bf16(aP, load_b_frag_t(&Vt[t2 * 16][0], 40), Oacc[t2]);
        }
    }

    // normalize and store (bf16) into [B*T][EMBED] at this head's columns
#pragma unroll
    for (int t2 = 0; t2 < 4; ++t2) {
#pragma unroll
        for (int v = 0; v < 8; ++v) {
            const int row = q0 + v + 8 * hi;
            const int col = h * HDIM + t2 * 16 + n;
            attn_out[((size_t)bb * TSEQ + row) * EMBED + col] =
                (bf16_t)(Oacc[t2][v] / lrow[v]);
        }
    }
}

// ---------------- host-side launcher ----------------
extern "C" void kernel_launch(void* const* d_in, const int* in_sizes, int n_in,
                              void* d_out, int out_size, void* d_ws, size_t ws_size,
                              hipStream_t stream) {
    (void)in_sizes; (void)n_in; (void)out_size; (void)ws_size;
    const float* x     = (const float*)d_in[0];
    const float* ln1_g = (const float*)d_in[1];
    const float* ln1_b = (const float*)d_in[2];
    const float* qkv_w = (const float*)d_in[3];
    const float* qkv_b = (const float*)d_in[4];
    const float* out_w = (const float*)d_in[5];
    const float* out_b = (const float*)d_in[6];
    const float* ln2_g = (const float*)d_in[7];
    const float* ln2_b = (const float*)d_in[8];
    const float* ff1_w = (const float*)d_in[9];
    const float* ff1_b = (const float*)d_in[10];
    const float* ff2_w = (const float*)d_in[11];
    const float* ff2_b = (const float*)d_in[12];

    char* ws = (char*)d_ws;
    size_t off = 0;
    auto walloc = [&](size_t bytes) -> char* {
        char* p = ws + off;
        off += (bytes + 255) & ~(size_t)255;
        return p;
    };
    bf16_t* wqkvT  = (bf16_t*)walloc((size_t)3 * EMBED * EMBED * sizeof(bf16_t));
    bf16_t* woutT  = (bf16_t*)walloc((size_t)EMBED * EMBED * sizeof(bf16_t));
    bf16_t* wff1T  = (bf16_t*)walloc((size_t)EMBED * FFDIM * sizeof(bf16_t));
    bf16_t* wff2T  = (bf16_t*)walloc((size_t)FFDIM * EMBED * sizeof(bf16_t));
    bf16_t* h1_bf  = (bf16_t*)walloc((size_t)BTOK * EMBED * sizeof(bf16_t));
    bf16_t* qkv_bf = (bf16_t*)walloc((size_t)BTOK * 3 * EMBED * sizeof(bf16_t));
    bf16_t* att_bf = (bf16_t*)walloc((size_t)BTOK * EMBED * sizeof(bf16_t));
    float*  x1_f   = (float*)walloc((size_t)BTOK * EMBED * sizeof(float));
    bf16_t* h2_bf  = (bf16_t*)walloc((size_t)BTOK * EMBED * sizeof(bf16_t));
    bf16_t* ff1_bf = (bf16_t*)walloc((size_t)BTOK * FFDIM * sizeof(bf16_t));

    // 1) weight convert + transpose to bf16 Wt[N][K]
    auto cvtGrid = [](size_t total) { return dim3((unsigned)((total + 255) / 256)); };
    cvt_transpose_kernel<<<cvtGrid((size_t)EMBED * 3 * EMBED), 256, 0, stream>>>(
        qkv_w, wqkvT, EMBED, 3 * EMBED);
    cvt_transpose_kernel<<<cvtGrid((size_t)EMBED * EMBED), 256, 0, stream>>>(
        out_w, woutT, EMBED, EMBED);
    cvt_transpose_kernel<<<cvtGrid((size_t)EMBED * FFDIM), 256, 0, stream>>>(
        ff1_w, wff1T, EMBED, FFDIM);
    cvt_transpose_kernel<<<cvtGrid((size_t)FFDIM * EMBED), 256, 0, stream>>>(
        ff2_w, wff2T, FFDIM, EMBED);

    // 2) LN1 -> bf16
    ln_bf16_kernel<<<BTOK, 256, 0, stream>>>(x, ln1_g, ln1_b, h1_bf);

    // 3) qkv = h1 @ qkv_w + qkv_b  (bf16 out)
    gemm_bf16_wmma_kernel<<<dim3(3 * EMBED / 128, BTOK / 64), 256, 0, stream>>>(
        h1_bf, wqkvT, qkv_b, nullptr, qkv_bf, BTOK, 3 * EMBED, EMBED, 0);

    // 4) causal flash attention -> att_bf
    attention_wmma_kernel<<<dim3(TSEQ / 64, NHEADS, NBATCH), 128, 0, stream>>>(
        qkv_bf, att_bf);

    // 5) x1 = x + att @ out_w + out_b  (f32 out)
    gemm_bf16_wmma_kernel<<<dim3(EMBED / 128, BTOK / 64), 256, 0, stream>>>(
        att_bf, woutT, out_b, x, x1_f, BTOK, EMBED, EMBED, 2);

    // 6) LN2 -> bf16
    ln_bf16_kernel<<<BTOK, 256, 0, stream>>>(x1_f, ln2_g, ln2_b, h2_bf);

    // 7) ff1 = gelu(h2 @ ff1_w + ff1_b)  (bf16 out)
    gemm_bf16_wmma_kernel<<<dim3(FFDIM / 128, BTOK / 64), 256, 0, stream>>>(
        h2_bf, wff1T, ff1_b, nullptr, ff1_bf, BTOK, FFDIM, EMBED, 1);

    // 8) out = x1 + ff1 @ ff2_w + ff2_b  (f32 out)
    gemm_bf16_wmma_kernel<<<dim3(EMBED / 128, BTOK / 64), 256, 0, stream>>>(
        ff1_bf, wff2T, ff2_b, x1_f, d_out, BTOK, EMBED, FFDIM, 2);
}
